// MultiHeadSelfAttention_56684978372884
// MI455X (gfx1250) — matrix operand order repaired and checked by reading, not compiled, not run on hardware
//
#include <hip/hip_runtime.h>

// ---------- types ----------
typedef __attribute__((ext_vector_type(16))) __bf16       v16bf;
typedef __attribute__((ext_vector_type(8)))  float        v8f;
typedef __attribute__((ext_vector_type(4)))  unsigned int u32x4;
typedef __attribute__((ext_vector_type(2)))  unsigned int u32x2;
typedef __attribute__((ext_vector_type(2)))  float        f32x2;

union FragCast {
    struct { u32x4 lo, hi; } b;
    v16bf v;
};

__device__ __forceinline__ unsigned short f2bf(float f) {
    unsigned u = __float_as_uint(f);
    u += 0x7fffu + ((u >> 16) & 1u);        // round-to-nearest-even
    return (unsigned short)(u >> 16);
}
__device__ __forceinline__ float bfhi2f(unsigned w) { return __uint_as_float(w & 0xffff0000u); }
__device__ __forceinline__ float bflo2f(unsigned w) { return __uint_as_float(w << 16); }

// 16-byte async copy global -> LDS (GLOBAL_LOAD_ASYNC_TO_LDS_B128, ASYNCcnt)
__device__ __forceinline__ void async_cp16(const void* g, unsigned lds_off) {
    asm volatile("global_load_async_to_lds_b128 %0, %1, off"
                 :: "v"(lds_off), "v"(g)
                 : "memory");
}
__device__ __forceinline__ void async_wait0() {
    asm volatile("s_wait_asynccnt 0x0" ::: "memory");
}
__device__ __forceinline__ void async_wait6() {
    asm volatile("s_wait_asynccnt 0x6" ::: "memory");
}

// lane-xor swizzle (ds_swizzle: offset[14:10]=xor, [9:5]=or, [4:0]=and)
template <int XORM>
__device__ __forceinline__ float swz_xor(float v) {
    int r = __builtin_amdgcn_ds_swizzle(__float_as_int(v), (XORM << 10) | 0x1f);
    return __int_as_float(r);
}

// ---------- fp32 -> bf16 cast ----------
__global__ void cvt_bf16(const float* __restrict__ in, unsigned short* __restrict__ out, int n2) {
    int i = blockIdx.x * blockDim.x + threadIdx.x;
    int stride = gridDim.x * blockDim.x;
    const f32x2* in2 = (const f32x2*)in;
    unsigned* out2 = (unsigned*)out;
    for (; i < n2; i += stride) {
        f32x2 f = in2[i];
        out2[i] = (unsigned)f2bf(f.x) | ((unsigned)f2bf(f.y) << 16);
    }
}

// ---------- NT GEMM: Y[M,N] = A[M,K](bf16) * Bw[N,K](bf16)^T ----------
// Block tile 128x256 (8 waves, each 64x64 = 4x4 wmma tiles), K step 32.
// A (8KB) and B (16KB) tiles double-buffered in LDS via async copies.
template <bool OUT_F32>
__global__ __launch_bounds__(256) void gemm_nt(const unsigned short* __restrict__ A,
                                               const unsigned short* __restrict__ Bw,
                                               void* __restrict__ Y,
                                               int M, int N, int K) {
    __shared__ unsigned short Abuf[2][128 * 32];
    __shared__ unsigned short Bbuf[2][256 * 32];

    const int th = threadIdx.x;
    const int lane = th & 31;
    const int wave = th >> 5;
    const int bM0 = blockIdx.y * 128;
    const int bN0 = blockIdx.x * 256;
    const int wm = (wave & 1) * 64;
    const int wn = (wave >> 1) * 64;
    const int row = lane & 15;
    const int hlf = lane >> 4;

    // cooperative async staging: A = 512 chunks of 16B, B = 1024 chunks of 16B
    auto stage = [&](int buf, int k0) {
        unsigned la = (unsigned)(uintptr_t)&Abuf[buf][0];
        unsigned lb = (unsigned)(uintptr_t)&Bbuf[buf][0];
#pragma unroll
        for (int s = 0; s < 2; ++s) {
            int c = th + 256 * s;
            int r = c >> 2, ko = c & 3;
            async_cp16(A + (size_t)(bM0 + r) * K + k0 + ko * 8,
                       la + (unsigned)(r * 64 + ko * 16));
        }
#pragma unroll
        for (int s = 0; s < 4; ++s) {
            int c = th + 256 * s;
            int r = c >> 2, ko = c & 3;
            async_cp16(Bw + (size_t)(bN0 + r) * K + k0 + ko * 8,
                       lb + (unsigned)(r * 64 + ko * 16));
        }
    };

    v8f acc[4][4];
    const v8f vzero = {0.f, 0.f, 0.f, 0.f, 0.f, 0.f, 0.f, 0.f};
#pragma unroll
    for (int i = 0; i < 4; ++i)
#pragma unroll
        for (int j = 0; j < 4; ++j) acc[i][j] = vzero;

    stage(0, 0);
    const int KT = K / 32;
    for (int kt = 0; kt < KT; ++kt) {
        const int buf = kt & 1;
        if (kt + 1 < KT) {
            stage(buf ^ 1, (kt + 1) * 32);   // 6 new outstanding async ops
            async_wait6();                   // current tile's 6 have completed
        } else {
            async_wait0();
        }
        __syncthreads();

        v16bf af[4], bfm[4];
#pragma unroll
        for (int i = 0; i < 4; ++i) {
            // A frag: lanes 0-15 K=0..7 (v0-3) & 16..23 (v4-7); lanes 16-31 +8
            const unsigned short* p = &Abuf[buf][(wm + i * 16 + row) * 32 + 8 * hlf];
            FragCast u;
            u.b.lo = *(const u32x4*)p;
            u.b.hi = *(const u32x4*)(p + 16);
            af[i] = u.v;
        }
#pragma unroll
        for (int j = 0; j < 4; ++j) {
            // B frag: lane n(+16h) holds col n, K=8h..8h+7 (v0-3) & 16+8h..23+8h (v4-7)
            const unsigned short* p = &Bbuf[buf][(wn + j * 16 + row) * 32 + 8 * hlf];
            FragCast u;
            u.b.lo = *(const u32x4*)p;
            u.b.hi = *(const u32x4*)(p + 16);
            bfm[j] = u.v;
        }
#pragma unroll
        for (int i = 0; i < 4; ++i)
#pragma unroll
            for (int j = 0; j < 4; ++j)
                acc[i][j] = __builtin_amdgcn_wmma_f32_16x16x32_bf16(
                    false, af[i], false, bfm[j], (short)0, acc[i][j], false, false);

        __syncthreads();
    }

    // C/D layout: VGPR r holds M=r (lanes 0-15) / M=r+8 (lanes 16-31), N=lane&15
#pragma unroll
    for (int i = 0; i < 4; ++i)
#pragma unroll
        for (int j = 0; j < 4; ++j)
#pragma unroll
            for (int r = 0; r < 8; ++r) {
                int m = bM0 + wm + i * 16 + r + 8 * hlf;
                int n = bN0 + wn + j * 16 + row;
                if constexpr (OUT_F32)
                    ((float*)Y)[(size_t)m * N + n] = acc[i][j][r];
                else
                    ((unsigned short*)Y)[(size_t)m * N + n] = f2bf(acc[i][j][r]);
            }
}

// ---------- per-token attention: scores = QK^T/sqrt(128), softmax, AV ----------
__global__ __launch_bounds__(256) void attn_tok(const unsigned short* __restrict__ Q,
                                                const unsigned short* __restrict__ Km,
                                                const unsigned short* __restrict__ V,
                                                unsigned short* __restrict__ O) {
    __shared__ float attn_s[8][16][17];
    __shared__ unsigned short vtile[8][2048];

    const int lane = threadIdx.x & 31;
    const int wave = threadIdx.x >> 5;
    const int t = blockIdx.x * 8 + wave;
    const int row = lane & 15;
    const int hlf = lane >> 4;

    const unsigned short* qp = Q + (size_t)t * 2048;
    const unsigned short* kp = Km + (size_t)t * 2048;

    // async-stage v rows 0..7 (8x128 bf16 = 2KB: 4 x 32 lanes x 16B) into LDS
    {
        const unsigned short* vp = V + (size_t)t * 2048;
        unsigned lv = (unsigned)(uintptr_t)&vtile[wave][0];
#pragma unroll
        for (int c = 0; c < 4; ++c) {
            int idx = lane + 32 * c;
            async_cp16(vp + idx * 8, lv + (unsigned)(idx * 16));
        }
    }

    // scores[h,g] = sum_d q[h,d]*k[g,d] : 4 chained bf16 WMMAs over d
    v8f s = {0.f, 0.f, 0.f, 0.f, 0.f, 0.f, 0.f, 0.f};
#pragma unroll
    for (int kk = 0; kk < 4; ++kk) {
        FragCast a, b;
        const unsigned short* pa = qp + row * 128 + kk * 32 + 8 * hlf;
        a.b.lo = *(const u32x4*)pa;
        a.b.hi = *(const u32x4*)(pa + 16);
        const unsigned short* pb = kp + row * 128 + kk * 32 + 8 * hlf;
        b.b.lo = *(const u32x4*)pb;
        b.b.hi = *(const u32x4*)(pb + 16);
        s = __builtin_amdgcn_wmma_f32_16x16x32_bf16(false, a.v, false, b.v, (short)0, s,
                                                    false, false);
    }

    // softmax over g (lanes 0..15 of each half hold one row's 16 columns)
    const float scale = 0.08838834764831845f;  // 1/sqrt(128)
#pragma unroll
    for (int r = 0; r < 8; ++r) {
        float xv = s[r] * scale;
        float mx = xv;
        mx = fmaxf(mx, swz_xor<1>(mx));
        mx = fmaxf(mx, swz_xor<2>(mx));
        mx = fmaxf(mx, swz_xor<4>(mx));
        mx = fmaxf(mx, swz_xor<8>(mx));
        float e = __expf(xv - mx);
        float sm = e;
        sm += swz_xor<1>(sm);
        sm += swz_xor<2>(sm);
        sm += swz_xor<4>(sm);
        sm += swz_xor<8>(sm);
        attn_s[wave][r + 8 * hlf][row] = e / sm;
    }

    // out[h,d] = sum_g attn[h,g] * v[g,d]; lane: h=lane&15, d-chunk of 64
    const int ho = row;
    const int d0 = hlf * 64;
    float ar[16];
#pragma unroll
    for (int g = 0; g < 16; ++g) ar[g] = attn_s[wave][ho][g];

    async_wait0();  // v tile landed in LDS (same-wave: wait orders async vs ds reads)

    unsigned short* op = O + (size_t)t * 2048 + ho * 128 + d0;
    for (int j4 = 0; j4 < 16; ++j4) {
        float a0 = 0.f, a1 = 0.f, a2 = 0.f, a3 = 0.f;
#pragma unroll
        for (int g = 0; g < 16; ++g) {
            u32x2 w = *(const u32x2*)&vtile[wave][g * 128 + d0 + j4 * 4];
            a0 += ar[g] * bflo2f(w.x);
            a1 += ar[g] * bfhi2f(w.x);
            a2 += ar[g] * bflo2f(w.y);
            a3 += ar[g] * bfhi2f(w.y);
        }
        u32x2 o2;
        o2.x = (unsigned)f2bf(a0) | ((unsigned)f2bf(a1) << 16);
        o2.y = (unsigned)f2bf(a2) | ((unsigned)f2bf(a3) << 16);
        *(u32x2*)(op + j4 * 4) = o2;
    }
}

// ---------- launch ----------
extern "C" void kernel_launch(void* const* d_in, const int* in_sizes, int n_in,
                              void* d_out, int out_size, void* d_ws, size_t ws_size,
                              hipStream_t stream) {
    (void)in_sizes; (void)n_in; (void)out_size; (void)ws_size;

    const float* x  = (const float*)d_in[0];   // [4,4096,2048]
    const float* Wq = (const float*)d_in[1];   // [2048,2048]
    const float* Wk = (const float*)d_in[2];
    const float* Wv = (const float*)d_in[3];
    const float* Wo = (const float*)d_in[4];

    const int M = 16384, N = 2048, K = 2048;
    const size_t NX = (size_t)M * K;           // 33,554,432
    const size_t NW = (size_t)N * K;           // 4,194,304

    unsigned short* wsb = (unsigned short*)d_ws;   // offsets in bf16 elements
    unsigned short* xb  = wsb;
    unsigned short* wqb = wsb + NX;
    unsigned short* wkb = wqb + NW;
    unsigned short* wvb = wkb + NW;
    unsigned short* wob = wvb + NW;
    unsigned short* qb  = wob + NW;
    unsigned short* kb  = qb + NX;
    unsigned short* vb  = kb + NX;
    unsigned short* ab  = vb + NX;             // attention output (bf16)

    // casts
    cvt_bf16<<<4096, 256, 0, stream>>>(x,  xb,  (int)(NX / 2));
    cvt_bf16<<<2048, 256, 0, stream>>>(Wq, wqb, (int)(NW / 2));
    cvt_bf16<<<2048, 256, 0, stream>>>(Wk, wkb, (int)(NW / 2));
    cvt_bf16<<<2048, 256, 0, stream>>>(Wv, wvb, (int)(NW / 2));
    cvt_bf16<<<2048, 256, 0, stream>>>(Wo, wob, (int)(NW / 2));

    // q, k, v projections
    dim3 g(N / 256, M / 128);
    gemm_nt<false><<<g, 256, 0, stream>>>(xb, wqb, qb, M, N, K);
    gemm_nt<false><<<g, 256, 0, stream>>>(xb, wkb, kb, M, N, K);
    gemm_nt<false><<<g, 256, 0, stream>>>(xb, wvb, vb, M, N, K);

    // per-token attention
    attn_tok<<<M / 8, 256, 0, stream>>>(qb, kb, vb, ab);

    // output projection -> fp32
    gemm_nt<true><<<g, 256, 0, stream>>>(ab, wob, d_out, M, N, K);
}
